// inference_model_45079976739346
// MI455X (gfx1250) — matrix operand order repaired and checked by reading, not compiled, not run on hardware
//
#include <hip/hip_runtime.h>
#include <hip/hip_bf16.h>
#include <math.h>

#define B_   16
#define EVI_ 5
#define L_   128
#define H_   768
#define K_   21
#define NL_  3

typedef __attribute__((ext_vector_type(16))) _Float16 v16h;
typedef __attribute__((ext_vector_type(8)))  float    v8f;
typedef __attribute__((ext_vector_type(4)))  unsigned u32x4;

union Frag16 { v16h v; unsigned u[8]; u32x4 q4[2]; };

#define SROW 72   // LDS row stride in halfs for a 64-half K-chunk: 144B, 16B-aligned

__constant__ float c_MU[K_] = {
  1.00f, 0.95f, 0.85f, 0.75f, 0.65f, 0.55f, 0.45f, 0.35f, 0.25f, 0.15f,
  0.05f,-0.05f,-0.15f,-0.25f,-0.35f,-0.45f,-0.55f,-0.65f,-0.75f,-0.85f,-0.95f };
__constant__ float c_IS2[K_] = {
  500000.0f, 50.f,50.f,50.f,50.f,50.f,50.f,50.f,50.f,50.f,
  50.f,50.f,50.f,50.f,50.f,50.f,50.f,50.f,50.f,50.f,50.f };

// ---------------- Stage A: l2-normalize rows, emit f16 ----------------
__global__ __launch_bounds__(256) void norm_kernel(const float* __restrict__ hid,
                                                   _Float16* __restrict__ hn) {
  const int row = blockIdx.x;                 // B*EVI*L rows of length H
  const int tid = threadIdx.x;
  const float* x = hid + (size_t)row * H_;
  float ss = 0.f;
  for (int i = tid; i < H_; i += 256) { float v = x[i]; ss += v * v; }
  __shared__ float r[256];
  r[tid] = ss; __syncthreads();
  for (int s = 128; s > 0; s >>= 1) {
    if (tid < s) r[tid] += r[tid + s];
    __syncthreads();
  }
  const float scale = 1.f / fmaxf(sqrtf(r[0]), 1e-12f);
  _Float16* o = hn + (size_t)row * H_;
  for (int i = tid; i < H_; i += 256) o[i] = (_Float16)(x[i] * scale);
}

// ---------------- Stage B: WMMA sim + fused kernel pooling ----------------
// grid = B*EVI*EVI blocks, block = 256 threads = 8 waves.
// wave ms computes rows q in [ms*16, ms*16+16) of the 128x128 sim tile.
// K streamed in 64-half chunks: 16 wmma per barrier pair.
struct SmemB {
  _Float16 q[L_ * SROW];    // K-chunk of Q (128x64 halfs, padded rows)  18432B
  _Float16 d[L_ * SROW];    // K-chunk of D                              18432B
  float    sim[L_ * 32];    // one 128x32 quarter of sim                 16384B
  float    red[L_ * K_];    // cross-thread pool combine                 10752B
  float    mcv[L_];
  float    selk[32];
};

__global__ __launch_bounds__(256) void sim_pool_kernel(
    const _Float16* __restrict__ hn,
    const float* __restrict__ mask_text,
    const float* __restrict__ mask_claim,
    const float* __restrict__ mask_evid,
    const float* __restrict__ W_att, const float* __restrict__ b_att,
    const float* __restrict__ W_sel, const float* __restrict__ b_sel,
    float* __restrict__ attLog,     // [B][eq][ed][L]
    float* __restrict__ selOut) {   // [B*EVI]
  __shared__ SmemB sm;
  const int blk = blockIdx.x;
  const int ed = blk % EVI_;
  const int eq = (blk / EVI_) % EVI_;
  const int b  = blk / (EVI_ * EVI_);
  const bool diag = (eq == ed);

  const int tid  = threadIdx.x;
  const int lane = tid & 31;
  const int wv   = tid >> 5;          // M-strip 0..7
  const int hi   = lane >> 4;         // half-wave
  const int ln   = lane & 15;

  const _Float16* Qg = hn + ((size_t)(b * EVI_ + eq) * L_) * H_;
  const _Float16* Dg = hn + ((size_t)(b * EVI_ + ed) * L_) * H_;

  v8f acc[8];
  #pragma unroll
  for (int ns = 0; ns < 8; ++ns) acc[ns] = (v8f){0.f,0.f,0.f,0.f,0.f,0.f,0.f,0.f};

  for (int kb = 0; kb < H_; kb += 64) {
    __syncthreads();
    // prefetch next K-chunk (one line per row per matrix)
    if (kb + 64 < H_) {
      const int prow = tid & 127;
      const _Float16* pp = (tid < 128) ? &Qg[(size_t)prow * H_ + kb + 64]
                                       : &Dg[(size_t)prow * H_ + kb + 64];
      __builtin_prefetch(pp, 0, 1);
    }
    // cooperative global->LDS copy of 128x64 f16 chunks (16B vectors)
    for (int p = tid; p < 1024; p += 256) {
      const int row = p >> 3;
      const int c8  = (p & 7) * 8;    // 16B-aligned half offset
      *(u32x4*)&sm.q[row * SROW + c8] = *(const u32x4*)&Qg[(size_t)row * H_ + kb + c8];
      *(u32x4*)&sm.d[row * SROW + c8] = *(const u32x4*)&Dg[(size_t)row * H_ + kb + c8];
    }
    __syncthreads();

    #pragma unroll
    for (int ks = 0; ks < 2; ++ks) {
      // A fragment: 16-bit A 16x32 layout (lanes0-15: K 0..7/16..23; lanes16-31: +8)
      Frag16 a;
      const int arow = (wv * 16 + ln) * SROW + ks * 32;
      a.q4[0] = *(const u32x4*)&sm.q[arow + hi * 8];
      a.q4[1] = *(const u32x4*)&sm.q[arow + 16 + hi * 8];

      // B fragments (same K order as A: lanes0-15 K 0..7/16..23; lanes16-31: +8; N=lane&15)
      Frag16 bf[8];
      #pragma unroll
      for (int ns = 0; ns < 8; ++ns) {
        const int brow = (ns * 16 + ln) * SROW + ks * 32;
        bf[ns].q4[0] = *(const u32x4*)&sm.d[brow + hi * 8];
        bf[ns].q4[1] = *(const u32x4*)&sm.d[brow + 16 + hi * 8];
      }
      #pragma unroll
      for (int ns = 0; ns < 8; ++ns) {
        acc[ns] = __builtin_amdgcn_wmma_f32_16x16x32_f16(
            false, a.v, false, bf[ns].v, (short)0, acc[ns], false, false);
      }
    }
  }

  // ---- fused Gaussian-kernel pooling over d, staged in four 128x32 quarters ----
  float pa[K_], ps[K_];
  #pragma unroll
  for (int k = 0; k < K_; ++k) { pa[k] = 0.f; ps[k] = 0.f; }

  const float* mtRow = mask_text + (size_t)(b * EVI_ + ed) * L_;
  const float* meRow = mask_evid + (size_t)(b * EVI_ + ed) * L_;
  const int qmine = tid >> 1;
  const int dsub  = (tid & 1) * 16;

  #pragma unroll
  for (int st = 0; st < 4; ++st) {
    __syncthreads();
    #pragma unroll
    for (int t = 0; t < 2; ++t) {
      const int ns = st * 2 + t;
      #pragma unroll
      for (int r = 0; r < 8; ++r) {
        const int qq = wv * 16 + hi * 8 + r;
        const int dd = ns * 16 + ln - st * 32;
        sm.sim[qq * 32 + dd] = acc[ns][r];
      }
    }
    __syncthreads();
    const int dbase = st * 32 + dsub;
    for (int j = 0; j < 16; ++j) {
      const int d = dbase + j;
      const float mt = mtRow[d];
      if (mt == 0.f) continue;              // mask_evid is a subset of mask_text
      const float me = diag ? meRow[d] : 0.f;
      const float s  = sm.sim[qmine * 32 + dsub + j];
      #pragma unroll
      for (int k = 0; k < K_; ++k) {
        const float df = s - c_MU[k];
        const float e  = __expf(-df * df * c_IS2[k]);
        pa[k] += e * mt;
        ps[k] += e * me;
      }
    }
  }

  // combine odd->even thread partials, project with W_att, mask, store logits
  __syncthreads();
  if (tid & 1) {
    #pragma unroll
    for (int k = 0; k < K_; ++k) sm.red[qmine * K_ + k] = pa[k];
  }
  __syncthreads();
  if (!(tid & 1)) {
    const int q = qmine;
    float sc = b_att[0];
    #pragma unroll
    for (int k = 0; k < K_; ++k) {
      const float pool = pa[k] + sm.red[q * K_ + k];
      sc += logf(fmaxf(pool, 1e-10f)) * W_att[k];
    }
    const float aq = mask_evid[(size_t)(b * EVI_ + eq) * L_ + q];
    attLog[((size_t)(b * EVI_ + eq) * EVI_ + ed) * L_ + q] = (aq == 0.f) ? -10000.f : sc;
  }

  // selection head only on diagonal blocks
  if (diag) {
    __syncthreads();
    if (tid & 1) {
      #pragma unroll
      for (int k = 0; k < K_; ++k) sm.red[qmine * K_ + k] = ps[k];
    }
    __syncthreads();
    if (!(tid & 1)) {
      const int q = qmine;
      const float mc = mask_claim[(size_t)(b * EVI_ + eq) * L_ + q];
      sm.mcv[q] = mc;
      #pragma unroll
      for (int k = 0; k < K_; ++k) {
        const float pool = ps[k] + sm.red[q * K_ + k];
        sm.red[q * K_ + k] = mc * 0.01f * logf(fmaxf(pool, 1e-10f));
      }
    }
    __syncthreads();
    if (tid < K_) {
      float s = 0.f;
      for (int q = 0; q < L_; ++q) s += sm.red[q * K_ + tid];
      sm.selk[tid] = s;
    }
    if (tid == 31) {
      float s = 0.f;
      for (int q = 0; q < L_; ++q) s += sm.mcv[q];
      sm.selk[K_] = s;
    }
    __syncthreads();
    if (tid == 0) {
      float num = 0.f;
      for (int k = 0; k < K_; ++k) num += sm.selk[k] * W_sel[k];
      selOut[b * EVI_ + eq] = num / (sm.selk[K_] + 1e-10f) + b_sel[0];
    }
  }
}

// ---------------- Stage C: att softmax + denoise weighted sum ----------------
// grid = EVI(i=ed) * B * EVI(eq)
__global__ __launch_bounds__(256) void denoise_kernel(
    const float* __restrict__ hid, const float* __restrict__ attLog,
    float* __restrict__ denoise) {
  const int blk = blockIdx.x;
  const int eq = blk % EVI_;
  const int b  = (blk / EVI_) % B_;
  const int ed = blk / (EVI_ * B_);
  const int tid = threadIdx.x;

  __shared__ float att[L_];
  const float* lg = attLog + ((size_t)(b * EVI_ + eq) * EVI_ + ed) * L_;
  if (tid < L_) att[tid] = lg[tid];
  __syncthreads();
  if (tid == 0) {
    float m = -1e30f;
    for (int q = 0; q < L_; ++q) m = fmaxf(m, att[q]);
    float s = 0.f;
    for (int q = 0; q < L_; ++q) { float e = __expf(att[q] - m); att[q] = e; s += e; }
    const float inv = 1.f / s;
    for (int q = 0; q < L_; ++q) att[q] *= inv;
  }
  __syncthreads();

  float o0 = 0.f, o1 = 0.f, o2 = 0.f;
  const float* base = hid + (size_t)(b * EVI_ + eq) * L_ * H_;
  for (int q = 0; q < L_; ++q) {
    const float a = att[q];
    const float* rowp = base + (size_t)q * H_;
    o0 += a * rowp[tid];
    o1 += a * rowp[tid + 256];
    o2 += a * rowp[tid + 512];
  }
  float* outp = denoise + ((size_t)(ed * B_ + b) * EVI_ + eq) * H_;
  outp[tid] = o0; outp[tid + 256] = o1; outp[tid + 512] = o2;
}

// ---------------- Stage D: GAT MLP + softmax over evidences ----------------
// grid = B*EVI (b,i)
__global__ __launch_bounds__(256) void gat_kernel(
    const float* __restrict__ inputs, const float* __restrict__ denoise,
    const float* __restrict__ W1, const float* __restrict__ b1,
    const float* __restrict__ W2, const float* __restrict__ b2,
    float* __restrict__ att_de) {
  const int blk = blockIdx.x;
  const int i = blk % EVI_, b = blk / EVI_;
  const int tid = threadIdx.x;
  __shared__ float red[128];
  __shared__ float g[EVI_];
  __shared__ float w[EVI_];
  const float* inp = inputs + (size_t)(b * EVI_ + i) * H_;

  for (int e = 0; e < EVI_; ++e) {
    const float* dn = denoise + ((size_t)(i * B_ + b) * EVI_ + e) * H_;
    if (tid < 128) {
      float acc = b1[tid];
      for (int h = 0; h < H_; ++h) acc += inp[h] * W1[h * 128 + tid];
      for (int h = 0; h < H_; ++h) acc += dn[h] * W1[(H_ + h) * 128 + tid];
      red[tid] = fmaxf(acc, 0.f) * W2[tid];
    }
    __syncthreads();
    if (tid == 0) {
      float s = b2[0];
      for (int j = 0; j < 128; ++j) s += red[j];
      g[e] = s;
    }
    __syncthreads();
  }
  if (tid == 0) {
    float m = -1e30f;
    for (int e = 0; e < EVI_; ++e) m = fmaxf(m, g[e]);
    float s = 0.f;
    for (int e = 0; e < EVI_; ++e) { w[e] = __expf(g[e] - m); s += w[e]; }
    for (int e = 0; e < EVI_; ++e) w[e] /= s;
  }
  __syncthreads();
  float o0 = 0.f, o1 = 0.f, o2 = 0.f;
  for (int e = 0; e < EVI_; ++e) {
    const float* dn = denoise + ((size_t)(i * B_ + b) * EVI_ + e) * H_;
    const float we = w[e];
    o0 += we * dn[tid];
    o1 += we * dn[tid + 256];
    o2 += we * dn[tid + 512];
  }
  float* outp = att_de + (size_t)(b * EVI_ + i) * H_;
  outp[tid] = o0; outp[tid + 256] = o1; outp[tid + 512] = o2;
}

// ---------------- Stage E: classifier + select mixture ----------------
__global__ __launch_bounds__(32) void final_kernel(
    const float* __restrict__ inputs, const float* __restrict__ att_de,
    const float* __restrict__ selArr,
    const float* __restrict__ inf_W, const float* __restrict__ inf_b,
    float* __restrict__ out) {
  const int b = blockIdx.x;
  const int tid = threadIdx.x;
  __shared__ float lg[EVI_][NL_];
  __shared__ float sv[EVI_];
  if (tid < EVI_ * NL_) {
    const int e = tid / NL_, c = tid % NL_;
    float s = inf_b[c];
    const float* ip = inputs + (size_t)(b * EVI_ + e) * H_;
    const float* ad = att_de + (size_t)(b * EVI_ + e) * H_;
    for (int h = 0; h < H_; ++h) s += ip[h] * inf_W[h * NL_ + c];
    for (int h = 0; h < H_; ++h) s += ad[h] * inf_W[(H_ + h) * NL_ + c];
    lg[e][c] = s;
  }
  if (tid >= 16 && tid < 16 + EVI_) sv[tid - 16] = selArr[b * EVI_ + (tid - 16)];
  __syncthreads();
  if (tid == 0) {
    float m = -1e30f, sp[EVI_];
    for (int e = 0; e < EVI_; ++e) m = fmaxf(m, sv[e]);
    float se = 0.f;
    for (int e = 0; e < EVI_; ++e) { sp[e] = __expf(sv[e] - m); se += sp[e]; }
    for (int e = 0; e < EVI_; ++e) sp[e] /= se;
    float res[NL_] = {0.f, 0.f, 0.f};
    for (int e = 0; e < EVI_; ++e) {
      float mm = fmaxf(fmaxf(lg[e][0], lg[e][1]), lg[e][2]);
      float cp[NL_]; float s2 = 0.f;
      for (int c = 0; c < NL_; ++c) { cp[c] = __expf(lg[e][c] - mm); s2 += cp[c]; }
      for (int c = 0; c < NL_; ++c) res[c] += sp[e] * cp[c] / s2;
    }
    for (int c = 0; c < NL_; ++c) out[b * NL_ + c] = logf(res[c]);
  }
}

extern "C" void kernel_launch(void* const* d_in, const int* in_sizes, int n_in,
                              void* d_out, int out_size, void* d_ws, size_t ws_size,
                              hipStream_t stream) {
  (void)in_sizes; (void)n_in; (void)out_size; (void)ws_size;
  const float* hid        = (const float*)d_in[0];
  const float* inputs     = (const float*)d_in[1];
  const float* mask_text  = (const float*)d_in[2];
  const float* mask_claim = (const float*)d_in[3];
  const float* mask_evid  = (const float*)d_in[4];
  const float* W_att = (const float*)d_in[5];
  const float* b_att = (const float*)d_in[6];
  const float* W_sel = (const float*)d_in[7];
  const float* b_sel = (const float*)d_in[8];
  const float* gW1 = (const float*)d_in[9];
  const float* gb1 = (const float*)d_in[10];
  const float* gW2 = (const float*)d_in[11];
  const float* gb2 = (const float*)d_in[12];
  const float* infW = (const float*)d_in[13];
  const float* infb = (const float*)d_in[14];
  float* out = (float*)d_out;

  char* ws = (char*)d_ws;
  _Float16* hn      = (_Float16*)(ws);                 // 10240*768*2 = 15,728,640 B
  float*    attLog  = (float*)(ws + 15728640);         // 16*5*5*128*4 = 204,800 B
  float*    selArr  = (float*)(ws + 15933440);         // 80*4 -> pad to 320
  float*    denoise = (float*)(ws + 15933760);         // 5*16*5*768*4 = 1,228,800 B
  float*    attDe   = (float*)(ws + 17162560);         // 16*5*768*4 = 245,760 B

  norm_kernel<<<B_ * EVI_ * L_, 256, 0, stream>>>(hid, hn);
  sim_pool_kernel<<<B_ * EVI_ * EVI_, 256, 0, stream>>>(
      hn, mask_text, mask_claim, mask_evid, W_att, b_att, W_sel, b_sel,
      attLog, selArr);
  denoise_kernel<<<EVI_ * B_ * EVI_, 256, 0, stream>>>(hid, attLog, denoise);
  gat_kernel<<<B_ * EVI_, 256, 0, stream>>>(inputs, denoise, gW1, gb1, gW2, gb2, attDe);
  final_kernel<<<B_, 32, 0, stream>>>(inputs, attDe, selArr, infW, infb, out);
}
